// OptimizedSelfONN_85220741087719
// MI455X (gfx1250) — matrix operand order repaired and checked by reading, not compiled, not run on hardware
//
#include <hip/hip_runtime.h>
#include <hip/hip_bf16.h>

typedef __attribute__((ext_vector_type(16))) __bf16 v16bf;
typedef __attribute__((ext_vector_type(8)))  float  v8f;

// pointer payload type for the async-to-LDS b128 builtin (from hipcc diagnostic):
// param0 = v4i in address_space(1) (global), param1 = LDS side
typedef int v4i_vs __attribute__((vector_size(16)));
typedef __attribute__((address_space(1))) v4i_vs g_v4i;
typedef __attribute__((address_space(3))) v4i_vs l_v4i;

#if __has_builtin(__builtin_amdgcn_global_load_async_to_lds_b128)
#define HAVE_ASYNC_LDS 1
#endif

__device__ __forceinline__ void sched_fence() {
#if __has_builtin(__builtin_amdgcn_sched_barrier)
    __builtin_amdgcn_sched_barrier(0);
#endif
}

__device__ __forceinline__ void wait_asynccnt0() {
#if __has_builtin(__builtin_amdgcn_s_wait_asynccnt)
    __builtin_amdgcn_s_wait_asynccnt(0);
#else
    asm volatile("s_wait_asynccnt 0x0" ::: "memory");
#endif
}

__device__ __forceinline__ float clipf(float v, float lo, float hi) {
    return fminf(fmaxf(v, lo), hi);
}

// ---------------- pack x: f32 -> clip(+-5) -> bf16 ----------------
__global__ void pack_x_kernel(const float* __restrict__ x, __bf16* __restrict__ xb, long n) {
    long i = ((long)blockIdx.x * blockDim.x + threadIdx.x) * 8;
    if (i + 8 > n) return;                    // n is a multiple of 8
    float4 f0 = *(const float4*)(x + i);
    float4 f1 = *(const float4*)(x + i + 4);
    union { __bf16 b[8]; uint4 u; } o;
    o.b[0] = (__bf16)clipf(f0.x, -5.0f, 5.0f);
    o.b[1] = (__bf16)clipf(f0.y, -5.0f, 5.0f);
    o.b[2] = (__bf16)clipf(f0.z, -5.0f, 5.0f);
    o.b[3] = (__bf16)clipf(f0.w, -5.0f, 5.0f);
    o.b[4] = (__bf16)clipf(f1.x, -5.0f, 5.0f);
    o.b[5] = (__bf16)clipf(f1.y, -5.0f, 5.0f);
    o.b[6] = (__bf16)clipf(f1.z, -5.0f, 5.0f);
    o.b[7] = (__bf16)clipf(f1.w, -5.0f, 5.0f);
    *(uint4*)(xb + i) = o.u;
}

// ------- pack weights: [O][I][Q=5] f32 -> [Q][O][I] bf16 (K-contiguous) -------
__global__ void pack_w_kernel(const float* __restrict__ w, __bf16* __restrict__ wb,
                              int O, int I) {
    long idx = (long)blockIdx.x * blockDim.x + threadIdx.x;   // over O*I (= o*I + i)
    long total = (long)O * I;
    if (idx >= total) return;
    const float* src = w + idx * 5;
    #pragma unroll
    for (int q = 0; q < 5; ++q)
        wb[(long)q * total + idx] = (__bf16)src[q];
}

// ---------------- softmax over operator_probs [O][5] ----------------
__global__ void softmax_ops_kernel(const float* __restrict__ p, float* __restrict__ opw, int O) {
    int o = blockIdx.x * blockDim.x + threadIdx.x;
    if (o >= O) return;
    float v[5];
    float m = -1e30f;
    #pragma unroll
    for (int q = 0; q < 5; ++q) { v[q] = p[o * 5 + q]; m = fmaxf(m, v[q]); }
    float s = 0.0f;
    #pragma unroll
    for (int q = 0; q < 5; ++q) { v[q] = __expf(v[q] - m); s += v[q]; }
    float inv = 1.0f / s;
    #pragma unroll
    for (int q = 0; q < 5; ++q) opw[o * 5 + q] = v[q] * inv;
}

// ---------------- fused GEMM (bf16 WMMA) + operator epilogue ----------------
#define BM 64
#define BN 32
#define BK 32
#define LDA 40        // padded LDS row stride (bf16 elems): 80B rows dodge bank conflicts
#define LDB 40
#define ASZ (BM * LDA)
#define BSZ (5 * BN * LDB)

__global__ __launch_bounds__(256, 1)
void onn_gemm_kernel(const __bf16* __restrict__ Xb,   // [B][I]  bf16 (pre-clipped)
                     const __bf16* __restrict__ Wb,   // [5][O][I] bf16
                     const float*  __restrict__ opw,  // [O][5] softmaxed
                     const float*  __restrict__ bias, // [O]
                     float*        __restrict__ out,  // [B][O]
                     int Bdim, int I, int O) {
    __shared__ __bf16 As[2 * ASZ];   // double-buffered
    __shared__ __bf16 Bs[2 * BSZ];

    const int tid  = threadIdx.x;
    const int lane = tid & 31;
    const int wv   = tid >> 5;                 // 8 waves = 4 (M) x 2 (N)
    const int wvm  = wv & 3;                   // M strip 0..3 (16 rows each)
    const int wvn  = wv >> 2;                  // N half  0..1 (16 cols each)
    const int o0   = blockIdx.x * BN;
    const int m0   = blockIdx.y * BM;

    v8f acc[5];
    #pragma unroll
    for (int q = 0; q < 5; ++q)
        #pragma unroll
        for (int e = 0; e < 8; ++e)
            acc[q][e] = 0.0f;

    // A-fill mapping: 256 threads -> 64 rows x 4 chunks of 8 bf16 (one b128 each)
    const int  ar   = tid >> 2;
    const int  ac   = (tid & 3) * 8;
    const bool aval = (m0 + ar) < Bdim;
    const long arow = (long)(m0 + ar);

    // B-fill mapping: threads 0..159 -> (q, n) rows of 32 bf16 (4 b128 each)
    const int  bq    = tid >> 5;               // 0..7, active when < 5
    const int  bn    = tid & 31;
    const long plane = (long)O * I;
    const bool bval  = (o0 + bn) < O;

#if defined(HAVE_ASYNC_LDS)
    // ---- CDNA5 async DMA fill: global -> LDS, tracked by ASYNCcnt ----
    auto fill_async = [&](int buf, int kk) {
        if (aval) {
            const __bf16* g = Xb + arow * I + kk + ac;
            __bf16* l = As + buf * ASZ + ar * LDA + ac;
            __builtin_amdgcn_global_load_async_to_lds_b128(
                (g_v4i*)(void*)(const_cast<__bf16*>(g)),
                (l_v4i*)(void*)l, 0, 0);
        }
        if (bq < 5 && bval) {
            const __bf16* g = Wb + (long)bq * plane + (long)(o0 + bn) * I + kk;
            __bf16* l = Bs + buf * BSZ + (bq * BN + bn) * LDB;
            #pragma unroll
            for (int c = 0; c < 4; ++c) {
                __builtin_amdgcn_global_load_async_to_lds_b128(
                    (g_v4i*)(void*)(const_cast<__bf16*>(g + c * 8)),
                    (l_v4i*)(void*)(l + c * 8), 0, 0);
            }
        }
    };
#else
    // ---- fallback: global -> registers -> LDS ----
    uint4 pa0, pb0, pb1, pb2, pb3;
    auto prefetch = [&](int kk) {
        if (aval) pa0 = *(const uint4*)(Xb + arow * I + kk + ac);
        else      pa0 = make_uint4(0u, 0u, 0u, 0u);
        if (bq < 5) {
            if (bval) {
                const __bf16* s = Wb + (long)bq * plane + (long)(o0 + bn) * I + kk;
                pb0 = *(const uint4*)(s);
                pb1 = *(const uint4*)(s + 8);
                pb2 = *(const uint4*)(s + 16);
                pb3 = *(const uint4*)(s + 24);
            } else {
                pb0 = make_uint4(0u, 0u, 0u, 0u); pb1 = pb0; pb2 = pb0; pb3 = pb0;
            }
        }
    };
    auto stage = [&](int buf) {
        *(uint4*)(As + buf * ASZ + ar * LDA + ac) = pa0;
        if (bq < 5) {
            __bf16* db = Bs + buf * BSZ + (bq * BN + bn) * LDB;
            *(uint4*)(db)      = pb0;
            *(uint4*)(db + 8)  = pb1;
            *(uint4*)(db + 16) = pb2;
            *(uint4*)(db + 24) = pb3;
        }
    };
#endif

    const int nk = I / BK;          // I is a multiple of 32

#if defined(HAVE_ASYNC_LDS)
    fill_async(0, 0);
    wait_asynccnt0();
    __syncthreads();
#else
    prefetch(0);
    stage(0);
    __syncthreads();
#endif

    for (int it = 0; it < nk; ++it) {
        const int cur = it & 1;

#if defined(HAVE_ASYNC_LDS)
        if (it + 1 < nk) fill_async(cur ^ 1, (it + 1) * BK);  // DMA next tile under compute
#else
        if (it + 1 < nk) prefetch((it + 1) * BK);
#endif

        // ---- A fragment: lane L -> row M=L%16; lanes<16: K 0..7 & 16..23, lanes>=16: +8 ----
        const __bf16* Ab = As + cur * ASZ;
        const __bf16* Bb = Bs + cur * BSZ;
        const int am = (wvm << 4) + (lane & 15);
        const int kh = (lane < 16) ? 0 : 8;
        v16bf afrag;
        {
            union { v16bf v; uint4 u[2]; } f;
            const __bf16* p = Ab + am * LDA;
            f.u[0] = *(const uint4*)(p + kh);        // K = kh .. kh+7
            f.u[1] = *(const uint4*)(p + kh + 16);   // K = kh+16 .. kh+23
            afrag = f.v;
        }
        // ---- load all 5 B fragments (one per operator plane) ----
        const int bnrow = (wvn << 4) + (lane & 15);     // column N within the tile
        const int bko   = (lane < 16) ? 0 : 8;          // K: kh..kh+7 & kh+16..kh+23
        v16bf bfr[5];
        #pragma unroll
        for (int q = 0; q < 5; ++q) {
            union { v16bf v; uint4 u[2]; } f;
            const __bf16* p = Bb + (q * BN + bnrow) * LDB + bko;
            f.u[0] = *(const uint4*)(p);
            f.u[1] = *(const uint4*)(p + 16);
            bfr[q] = f.v;
        }
        sched_fence();   // keep all ds_loads as one clause ahead of the WMMA burst
        #pragma unroll
        for (int q = 0; q < 5; ++q) {
            acc[q] = __builtin_amdgcn_wmma_f32_16x16x32_bf16(
                false, afrag, false, bfr[q], (short)0, acc[q], false, false);
        }

#if defined(HAVE_ASYNC_LDS)
        wait_asynccnt0();          // our DMA into the next buffer has landed
        __syncthreads();           // everyone's DMA has landed
#else
        if (it + 1 < nk) stage(cur ^ 1);
        __syncthreads();
#endif
    }

    // ---- epilogue: per element (b,o): mix 5 operator outputs ----
    const int col = o0 + (wvn << 4) + (lane & 15);
    if (col < O) {
        const float w0 = opw[col * 5 + 0];
        const float w1 = opw[col * 5 + 1];
        const float w2 = opw[col * 5 + 2];
        const float w3 = opw[col * 5 + 3];
        const float w4 = opw[col * 5 + 4];
        const float bz = bias[col];
        #pragma unroll
        for (int e = 0; e < 8; ++e) {
            const int row = m0 + (wvm << 4) + ((lane < 16) ? e : e + 8);
            if (row < Bdim) {
                float a0 = clipf(acc[0][e], -10.0f, 10.0f);
                float a1 = clipf(acc[1][e], -10.0f, 10.0f);
                float a2 = clipf(acc[2][e], -10.0f, 10.0f);
                float a3 = clipf(acc[3][e], -10.0f, 10.0f);
                float a4 = clipf(acc[4][e], -5.0f, 5.0f);   // inner clip(+-10) subsumed
                float e2 = __expf(2.0f * a3);
                float th = (e2 - 1.0f) / (e2 + 1.0f);       // tanh via exp (trans op)
                float r = a0 * w0 + __sinf(a1) * w1 + __cosf(a2) * w2
                        + th * w3 + __expf(a4) * w4 + bz;
                out[(long)row * O + col] = clipf(r, -10.0f, 10.0f);
            }
        }
    }
}

extern "C" void kernel_launch(void* const* d_in, const int* in_sizes, int n_in,
                              void* d_out, int out_size, void* d_ws, size_t ws_size,
                              hipStream_t stream) {
    const float* x     = (const float*)d_in[0];
    const float* w     = (const float*)d_in[1];
    const float* probs = (const float*)d_in[2];
    const float* bias  = (const float*)d_in[3];

    const int O = in_sizes[3];                 // 1024
    const int Q = in_sizes[2] / O;             // 5
    const int I = in_sizes[1] / (O * Q);       // 1024
    const int B = in_sizes[0] / I;             // 8192

    // workspace layout: [x bf16 | weights bf16 (Q planes) | op_w f32]  (~26 MB)
    __bf16* Xb  = (__bf16*)d_ws;
    __bf16* Wb  = Xb + (size_t)B * I;
    float*  opw = (float*)(Wb + (size_t)Q * O * I);

    const long nx = (long)B * I;
    pack_x_kernel<<<dim3((unsigned)((nx / 8 + 255) / 256)), dim3(256), 0, stream>>>(x, Xb, nx);
    const long nw = (long)O * I;
    pack_w_kernel<<<dim3((unsigned)((nw + 255) / 256)), dim3(256), 0, stream>>>(w, Wb, O, I);
    softmax_ops_kernel<<<dim3((O + 255) / 256), dim3(256), 0, stream>>>(probs, opw, O);

    dim3 grid((unsigned)(O / BN), (unsigned)((B + BM - 1) / BM));
    onn_gemm_kernel<<<grid, dim3(256), 0, stream>>>(Xb, Wb, opw, bias, (float*)d_out, B, I, O);

    (void)n_in; (void)out_size; (void)ws_size;
}
